// CfaModel_47717086658814
// MI455X (gfx1250) — matrix operand order repaired and checked by reading, not compiled, not run on hardware
//
#include <hip/hip_runtime.h>
#include <hip/hip_bf16.h>
#include <math.h>

// ---------------------------------------------------------------------------
// Problem constants (from reference)
// ---------------------------------------------------------------------------
#define BATCH   4
#define HH      64
#define PIX     (BATCH * HH * HH)      // 16384 pixels total
#define C_IN    1792
#define C_FEAT  1794                   // + xx, yy coords
#define C_PAD   1824                   // pad to multiple of 32 for WMMA K
#define C_OUT   448                    // multiple of 32 and of 16
#define KBANK   4096
#define NN      3

#if __has_builtin(__builtin_amdgcn_global_load_async_to_lds_b128)
#define HAVE_ASYNC_LDS 1
#else
#define HAVE_ASYNC_LDS 0
#endif
#if __has_builtin(__builtin_amdgcn_s_wait_asynccnt)
#define WAIT_ASYNC() __builtin_amdgcn_s_wait_asynccnt(0)
#else
#define WAIT_ASYNC() asm volatile("s_wait_asynccnt 0" ::: "memory")
#endif

typedef unsigned short ushort_t;
typedef __attribute__((ext_vector_type(16))) __bf16 v16bf;
typedef __attribute__((ext_vector_type(8)))  float  v8f;
typedef __attribute__((ext_vector_type(4)))  int    v4i;
typedef __attribute__((address_space(1))) v4i* gv4i_p;   // global-AS 16B vector ptr
typedef __attribute__((address_space(3))) v4i* lv4i_p;   // LDS-AS 16B vector ptr

union BF16Frag { v16bf v; uint4 q[2]; };

// ---------------------------------------------------------------------------
// bf16 helpers (round-to-nearest-even)
// ---------------------------------------------------------------------------
__device__ __forceinline__ ushort_t f2bf(float x) {
    unsigned u = __float_as_uint(x);
    u += 0x7FFFu + ((u >> 16) & 1u);
    return (ushort_t)(u >> 16);
}
__device__ __forceinline__ float bf2f(ushort_t h) {
    return __uint_as_float(((unsigned)h) << 16);
}

// ---------------------------------------------------------------------------
// WMMA fragment loaders (layouts per CDNA5 ISA §7.12.2, wave32)
// A (16x32 bf16): lane<16 -> row M=lane, elems {K0..7, K16..23};
//                 lane>=16 -> row M=lane-16, elems {K8..15, K24..31}
// B (32x16 bf16): lane<16 -> col N=lane,    elems {K0..7, K16..23};
//                 lane>=16 -> col N=lane-16, elems {K8..15, K24..31}
// ---------------------------------------------------------------------------
__device__ __forceinline__ v16bf load_row_frag(const ushort_t* __restrict__ base,
                                               int row0, int k0, int ld) {
    int lane = threadIdx.x & 31;
    int half = lane >> 4;
    int m    = row0 + (lane & 15);
    const ushort_t* p = base + (size_t)m * ld + k0 + half * 8;
    BF16Frag f;
    f.q[0] = *(const uint4*)(p);        // K +0..7   (upper half: +8..15)
    f.q[1] = *(const uint4*)(p + 16);   // K +16..23 (upper half: +24..31)
    return f.v;
}
// Same A layout, but sourced from an LDS-resident tile (rows 0..15) -> ds_load_b128
__device__ __forceinline__ v16bf load_row_frag_lds(const ushort_t* base, int k0, int ld) {
    int lane = threadIdx.x & 31;
    int half = lane >> 4;
    int m    = lane & 15;
    const ushort_t* p = base + m * ld + k0 + half * 8;
    BF16Frag f;
    f.q[0] = *(const uint4*)(p);
    f.q[1] = *(const uint4*)(p + 16);
    return f.v;
}
__device__ __forceinline__ v16bf load_col_frag(const ushort_t* __restrict__ base,
                                               int col0, int k0, int ld) {
    int lane = threadIdx.x & 31;
    int half = lane >> 4;
    int n    = col0 + (lane & 15);
    const ushort_t* p = base + (size_t)n * ld + k0 + half * 8;
    BF16Frag f;
    f.q[0] = *(const uint4*)(p);        // K +0..7   (upper half: +8..15)
    f.q[1] = *(const uint4*)(p + 16);   // K +16..23 (upper half: +24..31)
    return f.v;
}

// ---------------------------------------------------------------------------
// 3x3 avg pool, stride 1, pad 1, count_include_pad (sum/9)
// ---------------------------------------------------------------------------
__global__ void k_pool(const float* __restrict__ in, float* __restrict__ out,
                       int planes, int S) {
    int idx = blockIdx.x * blockDim.x + threadIdx.x;
    int total = planes * S * S;
    if (idx >= total) return;
    int x  = idx % S;
    int y  = (idx / S) % S;
    int pl = idx / (S * S);
    const float* img = in + (size_t)pl * S * S;
    float s = 0.f;
    #pragma unroll
    for (int dy = -1; dy <= 1; ++dy) {
        int yy = y + dy;
        if (yy < 0 || yy >= S) continue;
        #pragma unroll
        for (int dx = -1; dx <= 1; ++dx) {
            int xx = x + dx;
            if (xx < 0 || xx >= S) continue;
            s += img[yy * S + xx];
        }
    }
    out[idx] = s * (1.f / 9.f);
}

// bilinear sample (half-pixel centers, edge clamp) at dest (h,w), scale = Ssrc/64
__device__ __forceinline__ float bilin(const float* __restrict__ img, int S,
                                       int h, int w, float scale) {
    float sy = ((float)h + 0.5f) * scale - 0.5f;
    float sx = ((float)w + 0.5f) * scale - 0.5f;
    float fy = floorf(sy), fx = floorf(sx);
    float ty = sy - fy, tx = sx - fx;
    int y0 = (int)fy, x0 = (int)fx;
    int y1 = y0 + 1, x1 = x0 + 1;
    y0 = y0 < 0 ? 0 : y0;  x0 = x0 < 0 ? 0 : x0;
    y1 = y1 > S - 1 ? S - 1 : y1;  x1 = x1 > S - 1 ? S - 1 : x1;
    float v00 = img[y0 * S + x0], v01 = img[y0 * S + x1];
    float v10 = img[y1 * S + x0], v11 = img[y1 * S + x1];
    float a = v00 + tx * (v01 - v00);
    float c = v10 + tx * (v11 - v10);
    return a + ty * (c - a);
}

// ---------------------------------------------------------------------------
// Build bf16 feature matrix feat[pixel][C_PAD]
// ---------------------------------------------------------------------------
__global__ void k_feat(const float* __restrict__ pool1, const float* __restrict__ pool2,
                       const float* __restrict__ pool3, ushort_t* __restrict__ feat,
                       int total) {
    int idx = blockIdx.x * blockDim.x + threadIdx.x;
    if (idx >= total) return;
    int pix = idx / C_PAD;
    int c   = idx % C_PAD;
    int b = pix >> 12, p = pix & 4095, h = p >> 6, w = p & 63;
    float v;
    if (c < 256) {
        v = pool1[(((size_t)b * 256 + c) * 64 + h) * 64 + w];
    } else if (c < 768) {
        int cc = c - 256;
        v = bilin(pool2 + ((size_t)b * 512 + cc) * 32 * 32, 32, h, w, 0.5f);
    } else if (c < 1792) {
        int cc = c - 768;
        v = bilin(pool3 + ((size_t)b * 1024 + cc) * 16 * 16, 16, h, w, 0.25f);
    } else if (c == 1792) {
        v = ((float)h / 63.f) * 2.f - 1.f;   // xx varies along height
    } else if (c == 1793) {
        v = ((float)w / 63.f) * 2.f - 1.f;   // yy varies along width
    } else {
        v = 0.f;                              // K padding
    }
    feat[idx] = f2bf(v);
}

// W [C_OUT, C_FEAT] f32 -> Wb [C_OUT, C_PAD] bf16 (zero padded K)
__global__ void k_prep_w(const float* __restrict__ W, ushort_t* __restrict__ Wb, int total) {
    int idx = blockIdx.x * blockDim.x + threadIdx.x;
    if (idx >= total) return;
    int o = idx / C_PAD, c = idx % C_PAD;
    Wb[idx] = f2bf(c < C_FEAT ? W[(size_t)o * C_FEAT + c] : 0.f);
}

// memory_bank [C_OUT, KBANK] f32 -> mbT [KBANK, C_OUT] bf16 (transpose)
__global__ void k_prep_mb(const float* __restrict__ mb, ushort_t* __restrict__ mbT, int total) {
    int idx = blockIdx.x * blockDim.x + threadIdx.x;
    if (idx >= total) return;
    int k = idx / C_OUT, c = idx % C_OUT;
    mbT[idx] = f2bf(mb[(size_t)c * KBANK + k]);
}

// squared L2 norm of each bf16 row (one wave per row)
__global__ void k_norm(const ushort_t* __restrict__ X, float* __restrict__ nrm,
                       int rows, int ld) {
    int row = blockIdx.x * (blockDim.x >> 5) + (threadIdx.x >> 5);
    if (row >= rows) return;
    int lane = threadIdx.x & 31;
    float s = 0.f;
    for (int c = lane; c < ld; c += 32) {
        float v = bf2f(X[(size_t)row * ld + c]);
        s += v * v;
    }
    #pragma unroll
    for (int off = 16; off; off >>= 1) s += __shfl_down(s, off, 32);
    if (lane == 0) nrm[row] = s;
}

// ---------------------------------------------------------------------------
// GEMM1: t[pixel][o] = feat[pixel][:] . Wb[o][:] + bias[o]   (bf16 out)
// ---------------------------------------------------------------------------
__global__ void __launch_bounds__(256) k_gemm1(const ushort_t* __restrict__ feat,
                                               const ushort_t* __restrict__ Wb,
                                               const float* __restrict__ bias,
                                               ushort_t* __restrict__ tb) {
    int wave = (blockIdx.x * blockDim.x + threadIdx.x) >> 5;
    int mt = wave / (C_OUT / 16);      // pixel tile
    int nt = wave % (C_OUT / 16);      // out-channel tile
    int m0 = mt * 16, o0 = nt * 16;
    v8f acc = {};
    for (int k0 = 0; k0 < C_PAD; k0 += 32) {
        if (k0 + 32 < C_PAD) {
            __builtin_prefetch(feat + (size_t)m0 * C_PAD + k0 + 32, 0, 1);
            __builtin_prefetch(Wb + (size_t)o0 * C_PAD + k0 + 32, 0, 1);
        }
        v16bf a = load_row_frag(feat, m0, k0, C_PAD);
        v16bf b = load_col_frag(Wb, o0, k0, C_PAD);
        acc = __builtin_amdgcn_wmma_f32_16x16x32_bf16(false, a, false, b,
                                                      (short)0, acc, false, false);
    }
    int lane = threadIdx.x & 31, half = lane >> 4, col = o0 + (lane & 15);
    float bia = bias[col];
    #pragma unroll
    for (int r = 0; r < 8; ++r) {
        int row = m0 + r + 8 * half;   // C/D layout: VGPR r -> M=r (+8 for upper lanes)
        tb[(size_t)row * C_OUT + col] = f2bf(acc[r] + bia);
    }
}

// ---------------------------------------------------------------------------
// Fused distance + top-3 + softmin. One block = 16 pixels.
// The block's 16x448 bf16 descriptor tile is staged once into LDS via the
// CDNA5 async-to-LDS path (ASYNCcnt), then A fragments come from ds_load_b128
// while the 8 waves stream all 4096 memory-bank columns through WMMA.
// ---------------------------------------------------------------------------
__global__ void __launch_bounds__(256) k_dist(const ushort_t* __restrict__ tb,
                                              const ushort_t* __restrict__ mbT,
                                              const float* __restrict__ tnorm,
                                              const float* __restrict__ mnorm,
                                              float* __restrict__ out) {
    __shared__ __align__(16) ushort_t As[16 * C_OUT];   // 14 KB A tile
    __shared__ float tn_s[16];
    __shared__ float top_s[16 * 8 * 16 * NN];           // 24 KB candidate buffer
    int tid = threadIdx.x;
    int m0 = blockIdx.x * 16;
    if (tid < 16) tn_s[tid] = tnorm[m0 + tid];

    // Stage A tile: tb rows [m0, m0+16) are contiguous (16*448 bf16 = 896 x 16B)
    {
        const ushort_t* src = tb + (size_t)m0 * C_OUT;
#if HAVE_ASYNC_LDS
        for (int i = tid; i < (16 * C_OUT) / 8; i += 256) {
            __builtin_amdgcn_global_load_async_to_lds_b128(
                (gv4i_p)(src + i * 8),
                (lv4i_p)(As + i * 8),
                0, 0);
        }
        WAIT_ASYNC();
#else
        for (int i = tid; i < (16 * C_OUT) / 8; i += 256)
            *(uint4*)(As + i * 8) = *(const uint4*)(src + i * 8);
#endif
    }
    __syncthreads();

    int wave = tid >> 5, lane = tid & 31, half = lane >> 4, lc = lane & 15;
    float t0[8], t1[8], t2[8];
    #pragma unroll
    for (int r = 0; r < 8; ++r) { t0[r] = t1[r] = t2[r] = 3.4e38f; }

    for (int nt = wave; nt < KBANK / 16; nt += 8) {
        int col0 = nt * 16;
        if (nt + 8 < KBANK / 16)
            __builtin_prefetch(mbT + (size_t)(col0 + 128) * C_OUT, 0, 1);
        v8f acc = {};
        #pragma unroll 2
        for (int kc = 0; kc < C_OUT / 32; ++kc) {
            v16bf a = load_row_frag_lds(As, kc * 32, C_OUT);
            v16bf b = load_col_frag(mbT, col0, kc * 32, C_OUT);
            acc = __builtin_amdgcn_wmma_f32_16x16x32_bf16(false, a, false, b,
                                                          (short)0, acc, false, false);
        }
        float mn = mnorm[col0 + lc];
        #pragma unroll
        for (int r = 0; r < 8; ++r) {
            int row = r + 8 * half;
            float d2 = tn_s[row] + mn - 2.f * acc[r];
            float d = sqrtf(fmaxf(d2, 1e-12f));
            if (d < t0[r])      { t2[r] = t1[r]; t1[r] = t0[r]; t0[r] = d; }
            else if (d < t1[r]) { t2[r] = t1[r]; t1[r] = d; }
            else if (d < t2[r]) { t2[r] = d; }
        }
    }

    #pragma unroll
    for (int r = 0; r < 8; ++r) {
        int row = r + 8 * half;
        int base = (row * 8 + wave) * 16 * NN + lc * NN;
        top_s[base + 0] = t0[r];
        top_s[base + 1] = t1[r];
        top_s[base + 2] = t2[r];
    }
    __syncthreads();

    if (tid < 16) {
        const float* v = top_s + tid * 8 * 16 * NN;
        float b0 = 3.4e38f, b1 = 3.4e38f, b2 = 3.4e38f;
        for (int i = 0; i < 8 * 16 * NN; ++i) {
            float d = v[i];
            if (d < b0)      { b2 = b1; b1 = b0; b0 = d; }
            else if (d < b1) { b2 = b1; b1 = d; }
            else if (d < b2) { b2 = d; }
        }
        // softmin over ascending [b0,b1,b2]; weight of smallest * smallest
        float e1 = __expf(b0 - b1), e2 = __expf(b0 - b2);
        float sm0 = 1.f / (1.f + e1 + e2);
        out[m0 + tid] = sm0 * b0;
    }
}

// ---------------------------------------------------------------------------
// Host launcher
// ---------------------------------------------------------------------------
extern "C" void kernel_launch(void* const* d_in, const int* in_sizes, int n_in,
                              void* d_out, int out_size, void* d_ws, size_t ws_size,
                              hipStream_t stream) {
    const float* f1 = (const float*)d_in[0];   // [4,256,64,64]
    const float* f2 = (const float*)d_in[1];   // [4,512,32,32]
    const float* f3 = (const float*)d_in[2];   // [4,1024,16,16]
    const float* W  = (const float*)d_in[3];   // [448,1794]
    const float* bi = (const float*)d_in[4];   // [448]
    const float* mb = (const float*)d_in[5];   // [448,4096]
    float* out = (float*)d_out;                // [4,1,64,64] flat = 16384

    char* ws = (char*)d_ws;
    size_t off = 0;
    auto carve = [&](size_t bytes) -> void* {
        off = (off + 255) & ~(size_t)255;
        void* p = ws + off;
        off += bytes;
        return p;
    };
    float*    pool1 = (float*)carve((size_t)BATCH * 256 * 64 * 64 * 4);
    float*    pool2 = (float*)carve((size_t)BATCH * 512 * 32 * 32 * 4);
    float*    pool3 = (float*)carve((size_t)BATCH * 1024 * 16 * 16 * 4);
    ushort_t* feat  = (ushort_t*)carve((size_t)PIX * C_PAD * 2);
    ushort_t* Wb    = (ushort_t*)carve((size_t)C_OUT * C_PAD * 2);
    ushort_t* tb    = (ushort_t*)carve((size_t)PIX * C_OUT * 2);
    ushort_t* mbT   = (ushort_t*)carve((size_t)KBANK * C_OUT * 2);
    float*    tnrm  = (float*)carve((size_t)PIX * 4);
    float*    mnrm  = (float*)carve((size_t)KBANK * 4);
    (void)ws_size; (void)in_sizes; (void)n_in; (void)out_size;

    const int T = 256;
    {
        int tot = BATCH * 256 * 64 * 64;
        k_pool<<<(tot + T - 1) / T, T, 0, stream>>>(f1, pool1, BATCH * 256, 64);
    }
    {
        int tot = BATCH * 512 * 32 * 32;
        k_pool<<<(tot + T - 1) / T, T, 0, stream>>>(f2, pool2, BATCH * 512, 32);
    }
    {
        int tot = BATCH * 1024 * 16 * 16;
        k_pool<<<(tot + T - 1) / T, T, 0, stream>>>(f3, pool3, BATCH * 1024, 16);
    }
    {
        int tot = PIX * C_PAD;
        k_feat<<<(tot + T - 1) / T, T, 0, stream>>>(pool1, pool2, pool3, feat, tot);
    }
    {
        int tot = C_OUT * C_PAD;
        k_prep_w<<<(tot + T - 1) / T, T, 0, stream>>>(W, Wb, tot);
    }
    {
        int tot = KBANK * C_OUT;
        k_prep_mb<<<(tot + T - 1) / T, T, 0, stream>>>(mb, mbT, tot);
    }
    k_norm<<<(KBANK + 7) / 8, T, 0, stream>>>(mbT, mnrm, KBANK, C_OUT);
    {
        int waves = (PIX / 16) * (C_OUT / 16);
        k_gemm1<<<waves / 8, T, 0, stream>>>(feat, Wb, bi, tb);
    }
    k_norm<<<(PIX + 7) / 8, T, 0, stream>>>(tb, tnrm, PIX, C_OUT);
    k_dist<<<PIX / 16, T, 0, stream>>>(tb, mbT, tnrm, mnrm, out);
}
